// RCNNHead_16879221473475
// MI455X (gfx1250) — matrix operand order repaired and checked by reading, not compiled, not run on hardware
//
#include <hip/hip_runtime.h>
#include <hip/hip_bf16.h>
#include <math.h>

// ---------------------------------------------------------------------------
// R-CNN head for MI455X (gfx1250, wave32, WMMA).
// GEMMs: f16 WMMA (v_wmma_f32_16x16x32_f16), f32 accumulate.
// A tiles: async global->LDS copies (ASYNCcnt), double-buffered LDS.
// B fragments: double-buffered in registers across K-chunks so the inner
// WMMA loop waits only on DScnt.
// ---------------------------------------------------------------------------

typedef __attribute__((ext_vector_type(16))) _Float16 v16h;
typedef __attribute__((ext_vector_type(8)))  _Float16 v8h;
typedef __attribute__((ext_vector_type(8)))  float    v8f;
typedef int v4i __attribute__((vector_size(16)));   // matches builtin proto

#define KCONV 12544   // 256 channels * 7*7
#define NHID  1024
#define KC    128     // K halves staged per chunk
#define MT    32      // M rows per block
#define LDSS  136     // LDS row stride (halves): 68 dwords == 4 mod 64 banks

#if __has_builtin(__builtin_amdgcn_global_load_async_to_lds_b128)
#define USE_ASYNC 1
#else
#define USE_ASYNC 0
#endif

#if USE_ASYNC
__device__ __forceinline__ void async_b128(const _Float16* g, _Float16* l) {
  // gfx1250 GLOBAL_LOAD_ASYNC_TO_LDS_B128: 16B per lane, tracked by ASYNCcnt.
  __builtin_amdgcn_global_load_async_to_lds_b128((v4i*)g, (v4i*)l, 0, 0);
}
#if __has_builtin(__builtin_amdgcn_s_wait_asynccnt)
#define WAIT_ASYNC(n) __builtin_amdgcn_s_wait_asynccnt(n)
#else
#define WAIT_ASYNC(n) asm volatile("s_wait_asynccnt %0" ::"i"(n))
#endif
#endif

// ---------------- f32 -> f16 weight conversion ----------------
__global__ void f32_to_f16_kernel(const float* __restrict__ src,
                                  _Float16* __restrict__ dst, int n) {
  int i = blockIdx.x * blockDim.x + threadIdx.x;
  if (i < n) dst[i] = (_Float16)src[i];
}

// ---------------- fold BN (+conv bias) into scale/shift ----------------
__global__ void bn_fold_kernel(const float* __restrict__ gamma,
                               const float* __restrict__ beta,
                               const float* __restrict__ mean,
                               const float* __restrict__ var,
                               const float* __restrict__ conv_b,
                               float* __restrict__ s, float* __restrict__ t, int n) {
  int i = blockIdx.x * blockDim.x + threadIdx.x;
  if (i < n) {
    float sc = gamma[i] * rsqrtf(var[i] + 1e-3f);
    s[i] = sc;
    t[i] = beta[i] + (conv_b[i] - mean[i]) * sc;
  }
}

// ---------------- ROI-align, packed straight into GEMM-A (f16) ----------------
// One block per (padded) ROI row; 256 threads = 256 channels. Results staged in
// LDS and flushed with 16B vector stores.
__global__ __launch_bounds__(256)
void roi_align_pack_kernel(const float* __restrict__ feat2, const float* __restrict__ feat3,
                           const float* __restrict__ feat4, const float* __restrict__ feat5,
                           const float* __restrict__ rois, _Float16* __restrict__ Apack,
                           int R) {
  __shared__ __align__(16) _Float16 buf[KCONV];
  const int r = blockIdx.x;
  const int c = threadIdx.x;
  if (r >= R) {  // zero-fill pad rows so GEMM needs no M guards
    for (int p = 0; p < 49; ++p) buf[c * 49 + p] = (_Float16)0.f;
  } else {
    float x1 = rois[r * 4 + 0], y1 = rois[r * 4 + 1];
    float x2 = rois[r * 4 + 2], y2 = rois[r * 4 + 3];
    // FPN level: round(log2(sqrt(area)/224)) + 4, clipped to [2,5]
    float lvl = rintf(log2f(sqrtf((y2 - y1) * (x2 - x1)) * (1.f / 224.f))) + 4.f;
    lvl = fminf(fmaxf(lvl, 2.f), 5.f);
    const float* F; int H;
    if      (lvl < 2.5f) { F = feat2; H = 256; }
    else if (lvl < 3.5f) { F = feat3; H = 128; }
    else if (lvl < 4.5f) { F = feat4; H = 64;  }
    else                 { F = feat5; H = 32;  }
    const float inv_img = 1.f / 1024.f;
    float by1 = y1 * inv_img, bx1 = x1 * inv_img;
    float by2 = y2 * inv_img, bx2 = x2 * inv_img;
    float Hm1 = (float)(H - 1);
    const float* Fc = F + (size_t)c * H * H;
    for (int p = 0; p < 49; ++p) {
      int py = p / 7, px = p % 7;
      float ty = (float)py * (1.f / 6.f), tx = (float)px * (1.f / 6.f);
      float ys = (by1 + (by2 - by1) * ty) * Hm1;
      float xs = (bx1 + (bx2 - bx1) * tx) * Hm1;
      bool valid = (ys >= 0.f) && (ys <= Hm1) && (xs >= 0.f) && (xs <= Hm1);
      float yf = floorf(ys), xf = floorf(xs);
      float wy = ys - yf, wx = xs - xf;
      int y0  = (int)fminf(fmaxf(yf,       0.f), Hm1);
      int y1i = (int)fminf(fmaxf(yf + 1.f, 0.f), Hm1);
      int x0  = (int)fminf(fmaxf(xf,       0.f), Hm1);
      int x1i = (int)fminf(fmaxf(xf + 1.f, 0.f), Hm1);
      float v00 = Fc[y0  * H + x0], v01 = Fc[y0  * H + x1i];
      float v10 = Fc[y1i * H + x0], v11 = Fc[y1i * H + x1i];
      float val = v00 * (1.f - wy) * (1.f - wx) + v01 * (1.f - wy) * wx +
                  v10 * wy * (1.f - wx)         + v11 * wy * wx;
      buf[c * 49 + p] = (_Float16)(valid ? val : 0.f);
    }
  }
  __syncthreads();
  const v8h* s = (const v8h*)buf;
  v8h* d = (v8h*)(Apack + (size_t)r * KCONV);
  for (int i = threadIdx.x; i < KCONV / 8; i += 256) d[i] = s[i];
}

// ---------------- WMMA GEMM: C = epi(A[MxK] * W[NxK]^T) ----------------
// Block = 256 threads = 8 waves. Block tile 32(M) x 128(N); each wave owns two
// 16x16 M-subtiles sharing one B fragment (2 WMMAs per B load).
// EPI 0: x*p0[n] + p1[n]  (folded BN)    EPI 1: relu(x + p0[n])  (fc bias)
template <int EPI>
__global__ __launch_bounds__(256)
void wmma_gemm_kernel(const _Float16* __restrict__ A, const _Float16* __restrict__ W,
                      const float* __restrict__ p0, const float* __restrict__ p1,
                      _Float16* __restrict__ C, int N, int K) {
#if USE_ASYNC
  __shared__ __align__(16) _Float16 As[2][MT * LDSS];
#else
  __shared__ __align__(16) _Float16 As[1][MT * LDSS];
#endif
  const int tid  = threadIdx.x;
  const int lane = tid & 31;
  const int wave = tid >> 5;
  const int mbase = blockIdx.y * MT;
  const int ncol  = blockIdx.x * 128 + wave * 16 + (lane & 15);
  const int hi = lane >> 4;   // half-wave select (ISA 16-bit fragment layouts)
  const int m  = lane & 15;
  // staging map: 256 threads x 32B -> 32 rows x 256B (128 halves)
  const int srow = tid >> 3;
  const int scol = (tid & 7) * 16;
  const _Float16* gsrc = A + (size_t)(mbase + srow) * K + scol;
  _Float16* lds0 = &As[0][srow * LDSS + scol];
  const _Float16* Wrow = W + (size_t)ncol * K + hi * 16;

  v8f acc0 = {}, acc1 = {};
  const int nch = K / KC;

  // B fragments for one chunk, double-buffered in registers across chunks.
  v16h bf[4];
  auto loadB = [&](int c, v16h* dst) {
    const _Float16* wp = Wrow + (size_t)c * KC;
#pragma unroll
    for (int i = 0; i < 4; ++i) dst[i] = *(const v16h*)(wp + 32 * i);
  };
  auto compute_chunk = [&](const _Float16* buf, const v16h* b) {
#pragma unroll
    for (int ks = 0; ks < 4; ++ks) {
      const int ab = ks * 32 + hi * 16;
      // A frags: lane m holds K = {ab..ab+15} of rows m and m+16
      v8h a0lo = *(const v8h*)&buf[m * LDSS + ab];
      v8h a0hi = *(const v8h*)&buf[m * LDSS + ab + 8];
      v8h a1lo = *(const v8h*)&buf[(m + 16) * LDSS + ab];
      v8h a1hi = *(const v8h*)&buf[(m + 16) * LDSS + ab + 8];
      v16h a0, a1;
#pragma unroll
      for (int i = 0; i < 8; ++i) {
        a0[i] = a0lo[i]; a0[8 + i] = a0hi[i];
        a1[i] = a1lo[i]; a1[8 + i] = a1hi[i];
      }
      acc0 = __builtin_amdgcn_wmma_f32_16x16x32_f16(false, a0, false, b[ks],
                                                    (short)0, acc0, false, false);
      acc1 = __builtin_amdgcn_wmma_f32_16x16x32_f16(false, a1, false, b[ks],
                                                    (short)0, acc1, false, false);
    }
  };

  loadB(0, bf);
#if USE_ASYNC
  _Float16* lds1 = &As[1][srow * LDSS + scol];
  async_b128(gsrc,     lds0);      // prologue: chunk 0 -> buffer 0
  async_b128(gsrc + 8, lds0 + 8);
#endif
  for (int c = 0; c < nch; ++c) {
#if USE_ASYNC
    __syncthreads();               // chunk c-1 compute done -> (c+1)'s buffer free
    if (c + 1 < nch) {
      _Float16* ldsn = ((c + 1) & 1) ? lds1 : lds0;
      const _Float16* gn = gsrc + (size_t)(c + 1) * KC;
      async_b128(gn,     ldsn);
      async_b128(gn + 8, ldsn + 8);
      WAIT_ASYNC(2);               // our 2 copies for chunk c have landed
    } else {
      WAIT_ASYNC(0);
    }
    __syncthreads();               // every wave's part of chunk c landed
    const _Float16* buf = As[c & 1];
#else
    v16h st = *(const v16h*)(gsrc + (size_t)c * KC);
    __syncthreads();
    *(v16h*)lds0 = st;
    __syncthreads();
    const _Float16* buf = As[0];
#endif
    if (c + 1 < nch) {
      v16h bn[4];
      loadB(c + 1, bn);            // issue early: lands during chunk c compute
      compute_chunk(buf, bf);      // waits only on DScnt (B already in regs)
#pragma unroll
      for (int i = 0; i < 4; ++i) bf[i] = bn[i];
    } else {
      compute_chunk(buf, bf);
    }
  }
  // C layout (32-bit 16x16): VGPR v -> row v (lanes 0-15) / row 8+v (lanes 16-31)
  const float q0 = p0[ncol];
  const float q1 = (EPI == 0) ? p1[ncol] : 0.f;
#pragma unroll
  for (int v = 0; v < 8; ++v) {
    int mr = v + (hi ? 8 : 0);
    float x0 = acc0[v], x1 = acc1[v];
    if (EPI == 0) { x0 = x0 * q0 + q1;        x1 = x1 * q0 + q1; }
    else          { x0 = fmaxf(x0 + q0, 0.f); x1 = fmaxf(x1 + q0, 0.f); }
    C[(size_t)(mbase + mr)      * N + ncol] = (_Float16)x0;
    C[(size_t)(mbase + 16 + mr) * N + ncol] = (_Float16)x1;
  }
}

// ---------------- heads: cls logits + softmax + bbox ----------------
__global__ __launch_bounds__(256)
void head_kernel(const _Float16* __restrict__ H2,
                 const float* __restrict__ cls_w, const float* __restrict__ cls_b,
                 const float* __restrict__ bbox_w, const float* __restrict__ bbox_b,
                 float* __restrict__ out, int R) {
  __shared__ float red[256];
  __shared__ float res[10];
  const int r = blockIdx.x;
  const int t = threadIdx.x;
  const _Float16* h = H2 + (size_t)r * NHID;
  float hv[4];
#pragma unroll
  for (int i = 0; i < 4; ++i) hv[i] = (float)h[t + 256 * i];
  for (int o = 0; o < 10; ++o) {
    const float* w = (o < 2) ? (cls_w + o * NHID) : (bbox_w + (o - 2) * NHID);
    float p = 0.f;
#pragma unroll
    for (int i = 0; i < 4; ++i) p += hv[i] * w[t + 256 * i];
    red[t] = p;
    __syncthreads();
    for (int s = 128; s > 0; s >>= 1) {
      if (t < s) red[t] += red[t + s];
      __syncthreads();
    }
    if (t == 0) res[o] = red[0];
    __syncthreads();
  }
  if (t == 0) {
    float l0 = res[0] + cls_b[0], l1 = res[1] + cls_b[1];
    out[r * 2 + 0] = l0;
    out[r * 2 + 1] = l1;
    float mx = fmaxf(l0, l1);
    float e0 = expf(l0 - mx), e1 = expf(l1 - mx);
    float inv = 1.f / (e0 + e1);
    out[2 * R + r * 2 + 0] = e0 * inv;
    out[2 * R + r * 2 + 1] = e1 * inv;
#pragma unroll
    for (int o = 0; o < 8; ++o) out[4 * R + r * 8 + o] = res[2 + o] + bbox_b[o];
  }
}

// ---------------- host launch ----------------
extern "C" void kernel_launch(void* const* d_in, const int* in_sizes, int n_in,
                              void* d_out, int out_size, void* d_ws, size_t ws_size,
                              hipStream_t stream) {
  (void)n_in; (void)out_size; (void)ws_size;
  const float* feat2  = (const float*)d_in[0];
  const float* feat3  = (const float*)d_in[1];
  const float* feat4  = (const float*)d_in[2];
  const float* feat5  = (const float*)d_in[3];
  const float* rois   = (const float*)d_in[4];
  const float* conv_w = (const float*)d_in[5];
  const float* conv_b = (const float*)d_in[6];
  const float* bn_g   = (const float*)d_in[7];
  const float* bn_b   = (const float*)d_in[8];
  const float* bn_m   = (const float*)d_in[9];
  const float* bn_v   = (const float*)d_in[10];
  const float* fc1_w  = (const float*)d_in[11];
  const float* fc1_b  = (const float*)d_in[12];
  const float* fc2_w  = (const float*)d_in[13];
  const float* fc2_b  = (const float*)d_in[14];
  const float* cls_w  = (const float*)d_in[15];
  const float* cls_b  = (const float*)d_in[16];
  const float* bbox_w = (const float*)d_in[17];
  const float* bbox_b = (const float*)d_in[18];

  const int R = in_sizes[4] / 4;              // 1000
  const int Mpad = ((R + MT - 1) / MT) * MT;  // 1024

  char* ws = (char*)d_ws;
  size_t off = 0;
  auto carve = [&](size_t bytes) -> void* {
    void* p = ws + off;
    off += (bytes + 255) & ~(size_t)255;
    return p;
  };
  _Float16* Apack = (_Float16*)carve((size_t)Mpad * KCONV * sizeof(_Float16));
  _Float16* Wc    = (_Float16*)carve((size_t)NHID * KCONV * sizeof(_Float16));
  _Float16* W1    = (_Float16*)carve((size_t)NHID * NHID * sizeof(_Float16));
  _Float16* W2    = (_Float16*)carve((size_t)NHID * NHID * sizeof(_Float16));
  _Float16* Y0    = (_Float16*)carve((size_t)Mpad * NHID * sizeof(_Float16));
  _Float16* H1    = (_Float16*)carve((size_t)Mpad * NHID * sizeof(_Float16));
  _Float16* H2b   = (_Float16*)carve((size_t)Mpad * NHID * sizeof(_Float16));
  float*    bn_s  = (float*)carve(NHID * sizeof(float));
  float*    bn_t  = (float*)carve(NHID * sizeof(float));

  // weight conversion f32 -> f16
  const int nWc = NHID * KCONV;
  f32_to_f16_kernel<<<(nWc + 255) / 256, 256, 0, stream>>>(conv_w, Wc, nWc);
  const int nW1 = NHID * NHID;
  f32_to_f16_kernel<<<(nW1 + 255) / 256, 256, 0, stream>>>(fc1_w, W1, nW1);
  f32_to_f16_kernel<<<(nW1 + 255) / 256, 256, 0, stream>>>(fc2_w, W2, nW1);
  bn_fold_kernel<<<(NHID + 255) / 256, 256, 0, stream>>>(bn_g, bn_b, bn_m, bn_v,
                                                         conv_b, bn_s, bn_t, NHID);
  // ROI align into packed GEMM-A layout (pad rows zero-filled)
  roi_align_pack_kernel<<<Mpad, 256, 0, stream>>>(feat2, feat3, feat4, feat5,
                                                  rois, Apack, R);
  // GEMM chain (N = 1024 -> 8 N-tiles of 128; M tiles of 32)
  dim3 grid(NHID / 128, Mpad / MT);
  wmma_gemm_kernel<0><<<grid, 256, 0, stream>>>(Apack, Wc, bn_s, bn_t, Y0, NHID, KCONV);
  wmma_gemm_kernel<1><<<grid, 256, 0, stream>>>(Y0, W1, fc1_b, nullptr, H1, NHID, NHID);
  wmma_gemm_kernel<1><<<grid, 256, 0, stream>>>(H1, W2, fc2_b, nullptr, H2b, NHID, NHID);
  // heads
  head_kernel<<<R, 256, 0, stream>>>(H2b, cls_w, cls_b, bbox_w, bbox_b,
                                     (float*)d_out, R);
}
